// TPRU_11364483465529
// MI455X (gfx1250) — matrix operand order repaired and checked by reading, not compiled, not run on hardware
//
#include <hip/hip_runtime.h>
#include <hip/hip_bf16.h>
#include <stdint.h>

typedef __attribute__((ext_vector_type(16))) __bf16 v16bf;
typedef __attribute__((ext_vector_type(8)))  float  v8f;
typedef __attribute__((ext_vector_type(4)))  unsigned v4u;
typedef __attribute__((ext_vector_type(8)))  int      v8i;
typedef __attribute__((ext_vector_type(4)))  int      v4i;

#define BS   4096
#define DH   2048
#define NR   64
#define NSL  16
#define DSZ  128
#define BT   64            // rows per workgroup
#define NC   256           // cols per workgroup (128 f + 128 g)
#define KT   32            // K per stage (one WMMA K-step)
#define NKT  (DH / KT)     // 64 stages

// ---- LDS layout (byte offsets); all pitches multiples of 16 ----
#define APITCH 144         // 32 f32 row + 16B pad (TDM pad: 32 dw + 4 dw)
#define BPITCH 144
#define CPITCH 1040        // 256 f32 row + 16B pad
#define VPITCH 528         // 128 f32 row + 16B pad
#define XPITCH 528
#define OFF_A0   0
#define OFF_A1   9216
#define OFF_B0   18432
#define OFF_B1   55296     // stage region ends at 92160; C (0..66560) reuses it
#define OFF_VU   92160
#define OFF_VR   125952
#define OFF_XF   159744
#define OFF_XG   193536
#define OFF_A2   227328
#define OFF_RINV 243712
#define LDS_BYTES 243968

// async global -> LDS copy of 16 bytes (GVS mode: MEM[saddr + vaddr]) -> LDS[vdst]
__device__ __forceinline__ void async_b128(unsigned lds, unsigned voff,
                                           unsigned long long sbase) {
  asm volatile("global_load_async_to_lds_b128 %0, %1, %2"
               :: "v"(lds), "v"(voff), "s"(sbase) : "memory");
}

// Tensor Data Mover: 2-D tile (rows x 32 f32), row stride `stride_elems` f32,
// LDS gets 144-byte padded rows (pad_interval=4 -> every 32 DWORDs,
// pad_amount=3 -> 4 DWORDs).  D# per CDNA5 ISA.
__device__ __forceinline__ void tdm_load_2d(unsigned lds, unsigned long long gaddr,
                                            unsigned rows, unsigned stride_elems) {
  v4u g0;
  g0[0] = 1u;                                   // count=1, no gather/restore
  g0[1] = lds;                                  // lds_addr
  g0[2] = (unsigned)gaddr;                      // global_addr[31:0]
  g0[3] = ((unsigned)(gaddr >> 32) & 0x01FFFFFFu) | (2u << 30); // addr[56:32]|type=2
  v8i g1;
  g1[0] = (int)0x07120000u;   // data_size=4B, pad_enable, pad_interval=4, pad_amount=3
  g1[1] = (int)(32u << 16);   // tensor_dim0 = 32 (bits 79:48 low half)
  g1[2] = (int)(rows << 16);  // tensor_dim0 hi=0 | tensor_dim1 lo = rows
  g1[3] = (int)(32u << 16);   // tensor_dim1 hi=0 | tile_dim0 = 32
  g1[4] = (int)rows;          // tile_dim1 = rows, tile_dim2 = 0
  g1[5] = (int)stride_elems;  // tensor_dim0_stride[31:0]
  g1[6] = 0;                  // stride hi, tensor_dim1_stride lo
  g1[7] = 0;
  v4i z = {0, 0, 0, 0};
#if defined(__clang_major__) && (__clang_major__ >= 23)
  v8i e = {0, 0, 0, 0, 0, 0, 0, 0};
  __builtin_amdgcn_tensor_load_to_lds(g0, g1, z, z, e, 0);
#else
  __builtin_amdgcn_tensor_load_to_lds(g0, g1, z, z, 0);
#endif
}

// A operand (16x32 bf16, MxK): lane-half kh gets K = kh*8+0..7 (elems 0..7)
// and K = 16+kh*8+0..7 (elems 8..15).  r = f32 row base in LDS.
__device__ __forceinline__ v16bf make_opA(const float* r, int kh) {
  const float4* p0 = (const float4*)(r + kh * 8);
  const float4* p1 = (const float4*)(r + 16 + kh * 8);
  float4 a = p0[0], b = p0[1], c = p1[0], d = p1[1];
  v16bf o;
  o[0]=(__bf16)a.x; o[1]=(__bf16)a.y; o[2]=(__bf16)a.z; o[3]=(__bf16)a.w;
  o[4]=(__bf16)b.x; o[5]=(__bf16)b.y; o[6]=(__bf16)b.z; o[7]=(__bf16)b.w;
  o[8]=(__bf16)c.x; o[9]=(__bf16)c.y; o[10]=(__bf16)c.z; o[11]=(__bf16)c.w;
  o[12]=(__bf16)d.x; o[13]=(__bf16)d.y; o[14]=(__bf16)d.z; o[15]=(__bf16)d.w;
  return o;
}

// B operand (32x16 bf16, KxN): lane-half kh gets K = kh*8+0..7 (elems 0..7)
// and K = 16+kh*8+0..7 (elems 8..15).
__device__ __forceinline__ v16bf make_opB(const float* r, int kh) {
  const float4* p0 = (const float4*)(r + kh * 8);
  const float4* p1 = (const float4*)(r + 16 + kh * 8);
  float4 a = p0[0], b = p0[1], c = p1[0], d = p1[1];
  v16bf o;
  o[0]=(__bf16)a.x; o[1]=(__bf16)a.y; o[2]=(__bf16)a.z; o[3]=(__bf16)a.w;
  o[4]=(__bf16)b.x; o[5]=(__bf16)b.y; o[6]=(__bf16)b.z; o[7]=(__bf16)b.w;
  o[8]=(__bf16)c.x; o[9]=(__bf16)c.y; o[10]=(__bf16)c.z; o[11]=(__bf16)c.w;
  o[12]=(__bf16)d.x; o[13]=(__bf16)d.y; o[14]=(__bf16)d.z; o[15]=(__bf16)d.w;
  return o;
}

__global__ __launch_bounds__(256) void tpru_fused(
    const float* __restrict__ h,    const float* __restrict__ x,
    const float* __restrict__ v2ru, const float* __restrict__ W,
    const float* __restrict__ bfg,  const float* __restrict__ bhp,
    const float* __restrict__ bxp,  const float* __restrict__ bgp,
    float* __restrict__ out)
{
  extern __shared__ char smem[];
  const int t    = threadIdx.x;
  const int lane = t & 31;
  const int wave = t >> 5;
  const int s    = blockIdx.x;          // slice 0..15
  const int row0 = blockIdx.y * BT;     // batch-row tile base
  const unsigned sm = (unsigned)(uintptr_t)smem;  // LDS byte base (low 32 bits)

  // -------- one-shot async copies: VU, VR slices and x2f/x2g tiles ----------
  // 32 async b128 per thread; overlap the entire GEMM, waited before epilogue.
  {
    const unsigned long long vb = (unsigned long long)(uintptr_t)v2ru;
    const unsigned long long xb = (unsigned long long)(uintptr_t)x;
    #pragma unroll
    for (int j = 0; j < 8; ++j) {
      int c = t + j * 256;              // 0..2047 : 64 rows x 32 float4-chunks
      int r = c >> 5, dc = c & 31;
      async_b128(sm + OFF_VU + r * VPITCH + dc * 16,
                 (unsigned)(((r * NSL + s) * 2 * DSZ + DSZ + dc * 4) * 4), vb);
      async_b128(sm + OFF_VR + r * VPITCH + dc * 16,
                 (unsigned)(((r * NSL + s) * 2 * DSZ + dc * 4) * 4), vb);
      async_b128(sm + OFF_XF + r * XPITCH + dc * 16,
                 (unsigned)(((row0 + r) * 2 * DH + s * DSZ + dc * 4) * 4), xb);
      async_b128(sm + OFF_XG + r * XPITCH + dc * 16,
                 (unsigned)(((row0 + r) * 2 * DH + DH + s * DSZ + dc * 4) * 4), xb);
    }
  }

  const unsigned long long hb = (unsigned long long)(uintptr_t)h;
  const unsigned long long wb = (unsigned long long)(uintptr_t)W;

  // TDM stage: 3 tensor loads (A 64x32, Bf 128x32, Bg 128x32), wave 0 only.
  auto issue_stage = [&](int kt) {
    const int buf = kt & 1;
    const unsigned ab = sm + (buf ? OFF_A1 : OFF_A0);
    const unsigned bb = sm + (buf ? OFF_B1 : OFF_B0);
    const unsigned kb = (unsigned)(kt * KT);
    tdm_load_2d(ab, hb + (unsigned)(((unsigned)row0 * DH + kb) * 4), BT, DH);
    tdm_load_2d(bb, wb + (unsigned)((((unsigned)s * DSZ) * DH + kb) * 4), DSZ, DH);
    tdm_load_2d(bb + DSZ * BPITCH,
                wb + (unsigned)(((DH + (unsigned)s * DSZ) * DH + kb) * 4), DSZ, DH);
  };

  // -------- WMMA GEMM: C[64x256] = h_tile @ W_slice^T --------
  const int mt  = wave & 3;             // M-tile (16 rows)
  const int cgp = (wave >> 2) * 128;    // N-group base (8 N-tiles of 16)
  const int lm  = lane & 15;
  const int kh  = lane >> 4;
  v8f acc[8];
  #pragma unroll
  for (int i = 0; i < 8; ++i)
    #pragma unroll
    for (int j = 0; j < 8; ++j) acc[i][j] = 0.f;

  if (wave == 0) issue_stage(0);
  for (int kt = 0; kt < NKT; ++kt) {
    if (wave == 0) {
      if (kt + 1 < NKT) {
        issue_stage(kt + 1);
        __builtin_amdgcn_s_wait_tensorcnt((short)3);  // stage kt landed
      } else {
        __builtin_amdgcn_s_wait_tensorcnt((short)0);  // all stages landed
      }
    }
    __syncthreads();
    const char* Ab = smem + ((kt & 1) ? OFF_A1 : OFF_A0);
    const char* Bb = smem + ((kt & 1) ? OFF_B1 : OFF_B0);
    v16bf a = make_opA((const float*)(Ab + (mt * 16 + lm) * APITCH), kh);
    #pragma unroll
    for (int i = 0; i < 8; ++i) {
      v16bf b = make_opB((const float*)(Bb + (cgp + i * 16 + lm) * BPITCH), kh);
      acc[i] = __builtin_amdgcn_wmma_f32_16x16x32_bf16(
          false, a, false, b, (short)0, acc[i], false, false);
    }
    __syncthreads();                    // buf reusable next-next stage
  }

  // -------- store C (+ bias) to LDS (reuses stage region) --------
  #pragma unroll
  for (int i = 0; i < 8; ++i) {
    int col  = cgp + i * 16 + lm;       // 0..255  (0..127 = f, 128..255 = g)
    int wcol = (col < DSZ) ? (s * DSZ + col) : (DH + s * DSZ + (col - DSZ));
    float bias = bfg[wcol];
    #pragma unroll
    for (int r = 0; r < 8; ++r) {       // C layout: VGPR r -> M = r + 8*kh
      int row = mt * 16 + kh * 8 + r;
      *((float*)(smem + row * CPITCH) + col) = acc[i][r] + bias;
    }
  }
  asm volatile("s_wait_asynccnt 0" ::: "memory");  // VU/VR/XF/XG landed
  __syncthreads();

  const float bh0 = bhp[0], bx0 = bxp[0], bg0 = bgp[0];

  // -------- pass 1: a2[b][n] = (relu(x2f.vu + bh) + relu(h2f.vu + bx))^2 ----
  {
    int b  = t >> 2;
    int nb = (t & 3) * 16;
    const float* cf = (const float*)(smem + b * CPITCH);          // h2f row
    const float* xf = (const float*)(smem + OFF_XF + b * XPITCH); // x2f row
    float* a2p = (float*)(smem + OFF_A2);
    #pragma unroll 1
    for (int p = 0; p < 16; ++p) {
      int n = nb + p;
      const float* vu = (const float*)(smem + OFF_VU + n * VPITCH);
      float ax = 0.f, ah = 0.f;
      #pragma unroll 8
      for (int d = 0; d < DSZ; ++d) { float v = vu[d]; ax += cf[d] * v; ah += xf[d] * v; }
      ah = fmaxf(ah + bh0, 0.f);
      ax = fmaxf(ax + bx0, 0.f);
      float q = ah + ax;
      a2p[b * NR + n] = q * q;
    }
  }
  __syncthreads();

  // -------- pass 2: per-row L1 norm over roles --------
  if (t < BT) {
    const float* a2p = (const float*)(smem + OFF_A2) + t * NR;
    float sum = 0.f;
    #pragma unroll 8
    for (int n = 0; n < NR; ++n) sum += a2p[n];
    ((float*)(smem + OFF_RINV))[t] = 1.f / fmaxf(sum, 1e-12f);
  }
  __syncthreads();

  // -------- pass 3: hh = a @ vr ; out = g*hh + (1-g)*h --------
  {
    int b  = t >> 2;
    int db = (t & 3) * 32;
    float inv = ((const float*)(smem + OFF_RINV))[b];
    const float* a2p = (const float*)(smem + OFF_A2) + b * NR;
    float hh[32];
    #pragma unroll
    for (int d = 0; d < 32; ++d) hh[d] = 0.f;
    #pragma unroll 1
    for (int n = 0; n < NR; ++n) {
      float av = a2p[n] * inv;
      const float* vr = (const float*)(smem + OFF_VR + n * VPITCH) + db;
      #pragma unroll
      for (int d = 0; d < 32; ++d) hh[d] += av * vr[d];
    }
    const float* cgr = (const float*)(smem + b * CPITCH) + DSZ;     // h2g row
    const float* xg  = (const float*)(smem + OFF_XG + b * XPITCH);  // x2g row
    int gr = row0 + b;
    #pragma unroll
    for (int d = 0; d < 32; ++d) {
      int dd = db + d;
      float gs = xg[dd] + cgr[dd] + bg0;
      float gv = 1.f / (1.f + __expf(-gs));
      float hv = h[gr * DH + s * DSZ + dd];
      out[gr * DH + s * DSZ + dd] = gv * hh[d] + (1.f - gv) * hv;
    }
  }
}

extern "C" void kernel_launch(void* const* d_in, const int* in_sizes, int n_in,
                              void* d_out, int out_size, void* d_ws, size_t ws_size,
                              hipStream_t stream) {
  const float* h    = (const float*)d_in[0];
  const float* x    = (const float*)d_in[1];
  const float* v2ru = (const float*)d_in[2];
  const float* W    = (const float*)d_in[3];
  const float* bfg  = (const float*)d_in[4];
  const float* bh   = (const float*)d_in[5];
  const float* bx   = (const float*)d_in[6];
  const float* bg   = (const float*)d_in[7];
  (void)in_sizes; (void)n_in; (void)out_size; (void)d_ws; (void)ws_size;

  hipFuncSetAttribute((const void*)tpru_fused,
                      hipFuncAttributeMaxDynamicSharedMemorySize, LDS_BYTES);
  dim3 grid(NSL, BS / BT);   // 16 slices x 64 row-blocks
  dim3 block(256);           // 8 wave32
  tpru_fused<<<grid, block, LDS_BYTES, stream>>>(h, x, v2ru, W, bfg, bh, bx, bg,
                                                 (float*)d_out);
}
